// GroupedQueryAttention_51221779972724
// MI455X (gfx1250) — matrix operand order repaired and checked by reading, not compiled, not run on hardware
//
#include <hip/hip_runtime.h>
#include <stdint.h>

// ---------------------------------------------------------------------------
// GQA forward for MI455X (gfx1250, wave32): bf16 WMMA + TDM async staging +
// direct LDS row loads for K^T fragments.
// D=2048, HQ=16, HKV=4, head_dim=128, B=2, S=2048.
// ---------------------------------------------------------------------------

typedef __attribute__((ext_vector_type(16))) __bf16 v16bf;
typedef __attribute__((ext_vector_type(8)))  float  v8f;
typedef __attribute__((ext_vector_type(4)))  unsigned int v4u;
typedef __attribute__((ext_vector_type(8)))  int v8i_t;
typedef __attribute__((ext_vector_type(4)))  int v4i_t;

union FragAB {
  v16bf v;
  uint4 q[2];
};

__device__ __forceinline__ unsigned short f2bf(float f) {
  unsigned int u = __float_as_uint(f);
  unsigned int r = 0x7FFFu + ((u >> 16) & 1u);
  return (unsigned short)((u + r) >> 16);
}

__device__ __forceinline__ v8f wmma_bf16(const FragAB& a, const FragAB& b, v8f c) {
  // (neg_a, A, neg_b, B, c_mod, C, reuse_a, reuse_b)
  return __builtin_amdgcn_wmma_f32_16x16x32_bf16(false, a.v, false, b.v,
                                                 (short)0, c, false, false);
}

#define V8F_ZERO ((v8f){0.f,0.f,0.f,0.f,0.f,0.f,0.f,0.f})

// Low 32 bits of a generic shared pointer = LDS byte offset (flat->LDS mapping
// truncates to addr[31:0], ISA 10.2).
__device__ __forceinline__ unsigned lds_off(const void* p) {
  return (unsigned)(uintptr_t)p;
}

// ---------------------------------------------------------------------------
// TDM: async 2D tile load Global -> LDS (Tensor Data Mover, ISA ch. 7/8).
// 2D tile: tile_w elems wide (data_size=2B), tile_h rows, row stride in elems.
// D# packed per ISA 8.3/8.4; groups 2/3 zero (tile_dim2..4 unused).
// clang-23 toolchain: 6-arg builtin (g0, g1, g2, g3, g_extra, cpol).
// ---------------------------------------------------------------------------
__device__ __forceinline__ void tdm_load_2d_bf16(unsigned lds_byte_addr,
                                                 const void* gptr,
                                                 unsigned tile_w, unsigned tile_h,
                                                 unsigned row_stride) {
  unsigned long long ga = (unsigned long long)(uintptr_t)gptr;
  v4u g0;
  g0.x = 1u;                                               // count=1 (valid), user mode
  g0.y = lds_byte_addr;                                    // lds_addr [63:32]
  g0.z = (unsigned)(ga & 0xFFFFFFFFu);                     // global_addr [95:64]
  g0.w = (unsigned)((ga >> 32) & 0x1FFFFFFu) | (2u << 30); // addr[56:32] | type=2

  v8i_t g1;
  g1[0] = (int)(1u << 16);                                 // data_size=1 -> 2 bytes
  g1[1] = (int)((tile_w & 0xFFFFu) << 16);                 // tensor_dim0[15:0] @ [31:16]
  g1[2] = (int)(((tile_w >> 16) & 0xFFFFu) |               // tensor_dim0[31:16]
                ((tile_h & 0xFFFFu) << 16));               // tensor_dim1[15:0]
  g1[3] = (int)(((tile_h >> 16) & 0xFFFFu) |               // tensor_dim1[31:16]
                ((tile_w & 0xFFFFu) << 16));               // tile_dim0
  g1[4] = (int)(tile_h & 0xFFFFu);                         // tile_dim1 (tile_dim2=0)
  g1[5] = (int)row_stride;                                 // tensor_dim0_stride[31:0]
  g1[6] = 0;                                               // stride0[47:32], stride1[15:0]
  g1[7] = 0;                                               // stride1[47:16]

  v4i_t z4 = {0, 0, 0, 0};
  v8i_t z8 = {0, 0, 0, 0, 0, 0, 0, 0};
  __builtin_amdgcn_tensor_load_to_lds(g0, g1, z4, z4, z8, 0);
}

// ---------------------------------------------------------------------------
// fp32 -> bf16 conversion (grid-stride)
// ---------------------------------------------------------------------------
__global__ void __launch_bounds__(256)
cvt_f32_to_bf16(const float* __restrict__ src, unsigned short* __restrict__ dst, long n) {
  long i = (long)blockIdx.x * blockDim.x + threadIdx.x;
  long stride = (long)gridDim.x * blockDim.x;
  for (; i < n; i += stride) dst[i] = f2bf(src[i]);
}

// ---------------------------------------------------------------------------
// bf16 GEMM: C[M,N] = A[M,K] @ B[K,N], A/B row-major bf16, fp32 accumulate.
// Block tile 128x64, K-step 32, 256 threads = 8 waves, 1 wave = 16 rows x 64 cols.
// ---------------------------------------------------------------------------
#define BM 128
#define BN 64
#define BK 32
#define AST 40   // A LDS stride (bf16 elems), 80B rows keep 16B alignment
#define BST 72   // B LDS stride, 144B rows keep 16B alignment

template <bool OUT_BF16>
__global__ void __launch_bounds__(256)
gemm_bf16_wmma(const unsigned short* __restrict__ A,
               const unsigned short* __restrict__ B,
               void* __restrict__ Cv, int M, int N, int K) {
  __shared__ __align__(16) unsigned short sA[BM * AST];  // 10.0 KB
  __shared__ __align__(16) unsigned short sB[BK * BST];  //  4.5 KB

  const int tid  = threadIdx.x;
  const int lane = tid & 31;
  const int wave = tid >> 5;
  const int hlf  = lane >> 4;
  const int l16  = lane & 15;
  const int m0 = blockIdx.x * BM;
  const int n0 = blockIdx.y * BN;

  v8f acc[4];
#pragma unroll
  for (int j = 0; j < 4; ++j) acc[j] = V8F_ZERO;

  for (int k0 = 0; k0 < K; k0 += BK) {
    // stage A tile: 128x32 bf16 = 512 segments of 8 elems (16B)
#pragma unroll
    for (int i = 0; i < 2; ++i) {
      int s   = tid + i * 256;
      int row = s >> 2;            // 4 segs per row
      int off = (s & 3) * 8;
      *(uint4*)&sA[row * AST + off] =
          *(const uint4*)&A[(size_t)(m0 + row) * K + k0 + off];
    }
    // stage B tile: 32x64 bf16 = 256 segments
    {
      int row = tid >> 3;          // 8 segs per row
      int off = (tid & 7) * 8;
      *(uint4*)&sB[row * BST + off] =
          *(const uint4*)&B[(size_t)(k0 + row) * N + n0 + off];
    }
    // prefetch next K-step tiles into cache (global_prefetch_b8)
    if (k0 + BK < K) {
      __builtin_prefetch(&A[(size_t)(m0 + (tid >> 1)) * K + k0 + BK + (tid & 1) * 16], 0, 1);
      __builtin_prefetch(&B[(size_t)(k0 + BK + (tid >> 3)) * N + n0 + (tid & 7) * 8], 0, 1);
    }
    __syncthreads();

    FragAB a;
    const int arow = wave * 16 + l16;
    a.q[0] = *(const uint4*)&sA[arow * AST + hlf * 8];
    a.q[1] = *(const uint4*)&sA[arow * AST + 16 + hlf * 8];
#pragma unroll
    for (int j = 0; j < 4; ++j) {
      FragAB bf;
      for (int s = 0; s < 8; ++s) {
        ((unsigned short*)&bf)[s]     = sB[(hlf * 8 + s) * BST + j * 16 + l16];
        ((unsigned short*)&bf)[8 + s] = sB[(16 + hlf * 8 + s) * BST + j * 16 + l16];
      }
      acc[j] = wmma_bf16(a, bf, acc[j]);
    }
    __syncthreads();
  }

  // C/D layout: lane col = lane&15, rows = (lane>>4)*8 + r
  const int rbase = hlf * 8;
#pragma unroll
  for (int j = 0; j < 4; ++j) {
#pragma unroll
    for (int r = 0; r < 8; ++r) {
      int gr = m0 + wave * 16 + rbase + r;
      int gc = n0 + j * 16 + l16;
      float v = acc[j][r];
      if (OUT_BF16) ((unsigned short*)Cv)[(size_t)gr * N + gc] = f2bf(v);
      else          ((float*)Cv)[(size_t)gr * N + gc] = v;
    }
  }
}

// ---------------------------------------------------------------------------
// GQA flash attention (non-causal). Block = 128 threads = 4 waves.
// Each wave owns a 16-row Q tile; block covers 64 query rows of one (b,h).
// K and V tiles are both staged naturally (row-major) by the Tensor Data
// Mover (async, TENSORcnt). Score B-fragments (K^T) are produced with
// direct LDS row loads of K (lane = key, elems = dims); PV B-fragments read
// V rows directly.
// ---------------------------------------------------------------------------
#define S_LEN 2048
#define DH    128
#define PST   40    // sP stride: [qrow][key], 32 keys + pad

__global__ void __launch_bounds__(128)
gqa_attn_wmma(const unsigned short* __restrict__ Qb,
              const unsigned short* __restrict__ Kb,
              const unsigned short* __restrict__ Vb,
              unsigned short* __restrict__ Ob) {
  __shared__ __align__(16) unsigned short sKn[32 * DH];     // 8.0 KB (TDM dest, packed)
  __shared__ __align__(16) unsigned short sVt[32 * DH];     // 8.0 KB (TDM dest, packed)
  __shared__ __align__(16) unsigned short sP[4][16 * PST];  // 5.0 KB

  const int tid  = threadIdx.x;
  const int lane = tid & 31;
  const int wave = tid >> 5;
  const int hlf  = lane >> 4;
  const int l16  = lane & 15;

  const int bid = blockIdx.x;
  const int qb  = bid & 31;          // 32 q-blocks of 64 rows
  const int h   = (bid >> 5) & 15;   // query head
  const int b   = bid >> 9;          // batch
  const int kvh = h >> 2;            // GQA: 4 query heads per kv head

  const int qrow0 = qb * 64 + wave * 16;
  const unsigned sKn_base = lds_off(&sKn[0]);
  const unsigned sVt_base = lds_off(&sVt[0]);

  // Q fragments (A layout): lane row = l16; K runs [hlf*8,+8) and [16+hlf*8,+8)
  FragAB qf[4];
  {
    const unsigned short* qp =
        Qb + ((size_t)(b * S_LEN + qrow0 + l16) * 2048 + h * DH);
#pragma unroll
    for (int c = 0; c < 4; ++c) {
      qf[c].q[0] = *(const uint4*)(qp + c * 32 + hlf * 8);
      qf[c].q[1] = *(const uint4*)(qp + c * 32 + 16 + hlf * 8);
    }
  }

  v8f o[8];
#pragma unroll
  for (int d = 0; d < 8; ++d) o[d] = V8F_ZERO;
  float mrow[8], lrow[8];
#pragma unroll
  for (int r = 0; r < 8; ++r) { mrow[r] = -3.0e38f; lrow[r] = 0.f; }

  const float scale = 0.08838834764831845f;  // 1/sqrt(128)

  // TR16 tile addressing inside sKn[key][dim]: tile (keys f*16.., dims g*16..),
  // lane footprint = B128 footprint of the tile (lane%16 = tile row,
  // lane/16 selects the 16B column half).
  const unsigned trlane = sKn_base + (unsigned)(l16 * (DH * 2) + hlf * 16);

  for (int kt = 0; kt < S_LEN / 32; ++kt) {
    const int key0 = kt * 32;

    // ---- async TDM loads: K tile and V tile, [32 keys x 128 dims] each
    if (wave == 0) {
      tdm_load_2d_bf16(sKn_base,
                       Kb + ((size_t)(b * S_LEN + key0) * 512 + kvh * DH),
                       /*tile_w=*/DH, /*tile_h=*/32, /*row_stride=*/512);
      tdm_load_2d_bf16(sVt_base,
                       Vb + ((size_t)(b * S_LEN + key0) * 512 + kvh * DH),
                       /*tile_w=*/DH, /*tile_h=*/32, /*row_stride=*/512);
      __builtin_amdgcn_s_wait_tensorcnt(0);
    }
    __syncthreads();

    // ---- scores: S[16,32] = Q @ K^T, two 16-key column tiles.
    // B-fragments for K^T: lane = key, elems = dims (direct row loads).
    v8f s0 = V8F_ZERO, s1 = V8F_ZERO;
#pragma unroll
    for (int c = 0; c < 4; ++c) {
      FragAB k0f, k1f;
      k0f.q[0] = *(const uint4*)&sKn[l16 * DH + c * 32 + hlf * 8];
      k0f.q[1] = *(const uint4*)&sKn[l16 * DH + c * 32 + 16 + hlf * 8];
      k1f.q[0] = *(const uint4*)&sKn[(16 + l16) * DH + c * 32 + hlf * 8];
      k1f.q[1] = *(const uint4*)&sKn[(16 + l16) * DH + c * 32 + 16 + hlf * 8];
      s0 = wmma_bf16(qf[c], k0f, s0);
      s1 = wmma_bf16(qf[c], k1f, s1);
    }

    // ---- online softmax (rows split: lanes 0-15 rows 0-7, lanes 16-31 rows 8-15)
    float alpha[8];
#pragma unroll
    for (int r = 0; r < 8; ++r) {
      float a  = s0[r] * scale;   // (row, key = l16)
      float c2 = s1[r] * scale;   // (row, key = 16 + l16)
      float mx = fmaxf(a, c2);
#pragma unroll
      for (int off = 8; off >= 1; off >>= 1)
        mx = fmaxf(mx, __shfl_xor(mx, off, 32));
      float mnew = fmaxf(mrow[r], mx);
      float al = __expf(mrow[r] - mnew);
      float p0 = __expf(a - mnew);
      float p1 = __expf(c2 - mnew);
      float rs = p0 + p1;
#pragma unroll
      for (int off = 8; off >= 1; off >>= 1)
        rs += __shfl_xor(rs, off, 32);
      lrow[r] = lrow[r] * al + rs;
      mrow[r] = mnew;
      alpha[r] = al;
      const int prow = hlf * 8 + r;
      sP[wave][prow * PST + l16]      = f2bf(p0);
      sP[wave][prow * PST + 16 + l16] = f2bf(p1);
    }

#pragma unroll
    for (int d = 0; d < 8; ++d)
#pragma unroll
      for (int r = 0; r < 8; ++r) o[d][r] *= alpha[r];

    // per-wave LDS scratch: ensure P stores land before fragment reload
    asm volatile("s_wait_dscnt 0" ::: "memory");

    // ---- P fragment (A layout, 16x32)
    FragAB pf;
    pf.q[0] = *(const uint4*)&sP[wave][l16 * PST + hlf * 8];
    pf.q[1] = *(const uint4*)&sP[wave][l16 * PST + 16 + hlf * 8];

    // ---- O += P @ V : B-frag lane = head-dim col, elems = 16 key rows
#pragma unroll
    for (int d = 0; d < 8; ++d) {
      FragAB vf;
      for (int s = 0; s < 8; ++s) {
        ((unsigned short*)&vf)[s]     = sVt[(hlf * 8 + s) * DH + d * 16 + l16];
        ((unsigned short*)&vf)[8 + s] = sVt[(16 + hlf * 8 + s) * DH + d * 16 + l16];
      }
      o[d] = wmma_bf16(pf, vf, o[d]);
    }
    __syncthreads();
  }

  // ---- epilogue: O / l, write bf16 to [B*S, HQ*128]
  unsigned short* op = Ob + ((size_t)(b * S_LEN + qrow0) * 2048 + h * DH);
#pragma unroll
  for (int r = 0; r < 8; ++r) {
    float rinv = 1.0f / lrow[r];
    const int row = hlf * 8 + r;
#pragma unroll
    for (int d = 0; d < 8; ++d)
      op[(size_t)row * 2048 + d * 16 + l16] = f2bf(o[d][r] * rinv);
  }
}

// ---------------------------------------------------------------------------
// Host-side orchestration
// ---------------------------------------------------------------------------
extern "C" void kernel_launch(void* const* d_in, const int* in_sizes, int n_in,
                              void* d_out, int out_size, void* d_ws, size_t ws_size,
                              hipStream_t stream) {
  (void)in_sizes; (void)n_in; (void)out_size; (void)ws_size;
  const float* x  = (const float*)d_in[0];
  const float* Wq = (const float*)d_in[1];
  const float* Wk = (const float*)d_in[2];
  const float* Wv = (const float*)d_in[3];
  const float* Wo = (const float*)d_in[4];

  const int  Bb = 2, S = 2048, D = 2048, HQ = 16, HKV = 4;
  const long BS = (long)Bb * S;  // 4096

  unsigned short* ws = (unsigned short*)d_ws;
  unsigned short* xb  = ws; ws += BS * D;                 // 16 MB
  unsigned short* wqb = ws; ws += (long)D * (HQ  * 128);  //  8 MB
  unsigned short* wkb = ws; ws += (long)D * (HKV * 128);  //  2 MB
  unsigned short* wvb = ws; ws += (long)D * (HKV * 128);  //  2 MB
  unsigned short* wob = ws; ws += (long)D * D;            //  8 MB
  unsigned short* Qb  = ws; ws += BS * (HQ  * 128);       // 16 MB
  unsigned short* Kb  = ws; ws += BS * (HKV * 128);       //  4 MB
  unsigned short* Vb  = ws; ws += BS * (HKV * 128);       //  4 MB
  unsigned short* Ob  = ws;                               // 16 MB

  cvt_f32_to_bf16<<<2048, 256, 0, stream>>>(x,  xb,  BS * D);
  cvt_f32_to_bf16<<<2048, 256, 0, stream>>>(Wq, wqb, (long)D * HQ  * 128);
  cvt_f32_to_bf16<<<1024, 256, 0, stream>>>(Wk, wkb, (long)D * HKV * 128);
  cvt_f32_to_bf16<<<1024, 256, 0, stream>>>(Wv, wvb, (long)D * HKV * 128);
  cvt_f32_to_bf16<<<2048, 256, 0, stream>>>(Wo, wob, (long)D * D);

  // QKV projections (bf16 out)
  gemm_bf16_wmma<true><<<dim3((int)(BS / BM), 2048 / BN), 256, 0, stream>>>(
      xb, wqb, Qb, (int)BS, 2048, D);
  gemm_bf16_wmma<true><<<dim3((int)(BS / BM), 512 / BN), 256, 0, stream>>>(
      xb, wkb, Kb, (int)BS, 512, D);
  gemm_bf16_wmma<true><<<dim3((int)(BS / BM), 512 / BN), 256, 0, stream>>>(
      xb, wvb, Vb, (int)BS, 512, D);

  // attention: B * HQ * (S/64) blocks, 4 waves each
  gqa_attn_wmma<<<Bb * HQ * (S / 64), 128, 0, stream>>>(Qb, Kb, Vb, Ob);

  // out projection (fp32 out)
  gemm_bf16_wmma<false><<<dim3((int)(BS / BM), 2048 / BN), 256, 0, stream>>>(
      Ob, wob, d_out, (int)BS, 2048, D);
}
